// Linear4bitDUSQ_12403865551686
// MI455X (gfx1250) — 8-bit matrix operand order repaired and checked by reading, compiled, not run on hardware
//
#include <hip/hip_runtime.h>
#include <hip/hip_bf16.h>

typedef __attribute__((ext_vector_type(8))) int   v8i;
typedef __attribute__((ext_vector_type(8))) float v8f;

constexpr int T_DIM  = 2048;
constexpr int IN_DIM = 4096;
constexpr int OUT_DIM = 4096;
constexpr int KSTEPS = IN_DIM / 64;
constexpr int KSTEPS_HALF = KSTEPS / 2;
constexpr size_t X8_BYTES = (size_t)T_DIM * IN_DIM;
constexpr size_t W8_BYTES = (size_t)OUT_DIM * IN_DIM;

__global__ __launch_bounds__(256)
void pack_x_kernel(const int* __restrict__ xq, unsigned char* __restrict__ x8) {
  unsigned g    = blockIdx.x * 256u + threadIdx.x;
  unsigned tile = g >> 8;
  unsigned q    = g & 255u;
  unsigned lane = q >> 3;
  unsigned d    = q & 7u;
  unsigned tM = tile >> 6;
  unsigned tK = tile & 63u;
  unsigned M  = lane & 15u;
  unsigned K  = ((lane >> 4) << 3) + ((d >> 1) << 4) + ((d & 1u) << 2);
  const int* src = xq + (size_t)(tM * 16u + M) * IN_DIM + (tK * 64u + K);
  unsigned w = 0;
#pragma unroll
  for (int b = 0; b < 4; ++b) {
    int v = __builtin_nontemporal_load(src + b);
    w |= ((unsigned)(v & 0xFF)) << (8 * b);
  }
  reinterpret_cast<unsigned*>(x8)[g] = w;
}

__global__ __launch_bounds__(256)
void pack_w_kernel(const int* __restrict__ wp, unsigned char* __restrict__ w8) {
  unsigned g    = blockIdx.x * 256u + threadIdx.x;
  unsigned tile = g >> 8;
  unsigned q    = g & 255u;
  unsigned lane = q >> 3;
  unsigned d    = q & 7u;
  unsigned tN = tile >> 6;
  unsigned tK = tile & 63u;
  unsigned N  = lane & 15u;
  unsigned K  = ((lane >> 4) << 3) + ((d >> 1) << 4) + ((d & 1u) << 2);
  unsigned o  = tN * 16u + N;
  unsigned kc = tK * 64u + K;
  const int* src = wp + (size_t)o * (IN_DIM / 2) + (kc >> 1);
  int b0 = __builtin_nontemporal_load(src);
  int b1 = __builtin_nontemporal_load(src + 1);
  int nib[4] = { b0 & 0xF, (b0 >> 4) & 0xF, b1 & 0xF, (b1 >> 4) & 0xF };
  unsigned w = 0;
#pragma unroll
  for (int b = 0; b < 4; ++b) {
    int v = (nib[b] ^ 8) - 8;
    w |= ((unsigned)(v & 0xFF)) << (8 * b);
  }
  reinterpret_cast<unsigned*>(w8)[g] = w;
}

__global__ __launch_bounds__(256)
void gemm_i4_dusq_kernel(const unsigned char* __restrict__ x8,
                         const unsigned char* __restrict__ w8,
                         const float* __restrict__ sx,
                         const float* __restrict__ ws1,
                         const float* __restrict__ ws2,
                         const float* __restrict__ bias,
                         float* __restrict__ out) {
  const int lane  = threadIdx.x & 31;
  const int wave  = threadIdx.x >> 5;
  const int waveM = wave >> 2;
  const int waveN = wave & 3;
  const int wM = blockIdx.y * 64 + waveM * 32;
  const int wN = blockIdx.x * 128 + waveN * 32;

  const unsigned char* aB[2];
  const unsigned char* bB[2];
#pragma unroll
  for (int i = 0; i < 2; ++i) {
    aB[i] = x8 + ((size_t)((wM >> 4) + i) * KSTEPS) * 1024 + lane * 32;
    bB[i] = w8 + ((size_t)((wN >> 4) + i) * KSTEPS) * 1024 + lane * 32;
  }

  v8f facc[2][2] = {};

#pragma unroll
  for (int r = 0; r < 2; ++r) {
    v8i acc[2][2] = {};
    for (int ks = r * KSTEPS_HALF; ks < (r + 1) * KSTEPS_HALF; ++ks) {
      const size_t off  = (size_t)ks * 1024;
      const size_t poff = (size_t)(ks < KSTEPS - 1 ? ks + 1 : ks) * 1024;
      v8i a[2], b[2];
#pragma unroll
      for (int i = 0; i < 2; ++i) {
        a[i] = *reinterpret_cast<const v8i*>(aB[i] + off);
        b[i] = *reinterpret_cast<const v8i*>(bB[i] + off);
        __builtin_prefetch(aB[i] + poff, 0, 3);
        __builtin_prefetch(bB[i] + poff, 0, 3);
      }
#pragma unroll
      for (int mi = 0; mi < 2; ++mi)
#pragma unroll
        for (int ni = 0; ni < 2; ++ni)
          acc[mi][ni] = __builtin_amdgcn_wmma_i32_16x16x64_iu8(
               true, a[mi],  true, b[ni],
              acc[mi][ni],  false,  false);
    }
    const float* ws = r ? ws2 : ws1;
    float sc[2] = { ws[wN + (lane & 15)], ws[wN + 16 + (lane & 15)] };
#pragma unroll
    for (int mi = 0; mi < 2; ++mi)
#pragma unroll
      for (int ni = 0; ni < 2; ++ni)
#pragma unroll
        for (int v = 0; v < 8; ++v)
          facc[mi][ni][v] += (float)acc[mi][ni][v] * sc[ni];
  }

  const int n0 = wN + (lane & 15);
  const int n1 = n0 + 16;
  const float bv0 = bias[n0];
  const float bv1 = bias[n1];
  const int mrow = wM + ((lane >> 4) << 3);
#pragma unroll
  for (int mi = 0; mi < 2; ++mi)
#pragma unroll
    for (int v = 0; v < 8; ++v) {
      const int m = mrow + mi * 16 + v;
      const float s = sx[m];
      __builtin_nontemporal_store(s * facc[mi][0][v] + bv0,
                                  out + (size_t)m * OUT_DIM + n0);
      __builtin_nontemporal_store(s * facc[mi][1][v] + bv1,
                                  out + (size_t)m * OUT_DIM + n1);
    }
}

extern "C" void kernel_launch(void* const* d_in, const int* in_sizes, int n_in,
                              void* d_out, int out_size, void* d_ws, size_t ws_size,
                              hipStream_t stream) {
  (void)in_sizes; (void)n_in; (void)out_size; (void)ws_size;
  const int*   x_q  = (const int*)d_in[0];
  const float* sx   = (const float*)d_in[1];
  const int*   wpk  = (const int*)d_in[2];
  const float* ws1  = (const float*)d_in[3];
  const float* ws2  = (const float*)d_in[4];
  const float* bias = (const float*)d_in[5];
  float* out = (float*)d_out;

  unsigned char* x8 = (unsigned char*)d_ws;
  unsigned char* w8 = x8 + X8_BYTES;

  pack_x_kernel<<<(T_DIM * IN_DIM / 4) / 256, 256, 0, stream>>>(x_q, x8);
  pack_w_kernel<<<(OUT_DIM * IN_DIM / 4) / 256, 256, 0, stream>>>(wpk, w8);

  dim3 grid(OUT_DIM / 128, T_DIM / 64);
  gemm_i4_dusq_kernel<<<grid, 256, 0, stream>>>(x8, w8, sx, ws1, ws2, bias, out);
}
